// minGRU_46291157516471
// MI455X (gfx1250) — matrix operand order repaired and checked by reading, not compiled, not run on hardware
//
#include <hip/hip_runtime.h>

// Problem constants (from reference): B=4, S=8192, D=1024
#define B_  4
#define S_  8192
#define D_  1024
#define M_  (B_ * S_)      // 32768 rows of the GEMM
#define N2_ (2 * D_)       // 2048 cols of W

// GEMM tiling
#define TM  128            // M rows per workgroup
#define TNP 64             // hidden/gate column PAIRS per workgroup (=> 128 B-cols)
#define TK  32             // K per stage (one bf16 WMMA K)
#define XPITCH 40          // LDS row pitch in bf16 (80B: conflict-free, 16B-aligned)

// Scan tiling
#define CHUNK 256
#define NC (S_ / CHUNK)    // 32 chunks per channel

typedef __attribute__((ext_vector_type(4)))  __bf16 v4bf;
typedef __attribute__((ext_vector_type(8)))  __bf16 v8bf;
typedef __attribute__((ext_vector_type(16))) __bf16 v16bf;
typedef __attribute__((ext_vector_type(8)))  float  v8f;

#ifndef __has_builtin
#define __has_builtin(x) 0
#endif
#if __has_builtin(__builtin_amdgcn_tensor_load_to_lds) && \
    __has_builtin(__builtin_amdgcn_s_wait_tensorcnt)
#define USE_TDM 1
#else
#define USE_TDM 0
#endif

#if USE_TDM
typedef __attribute__((ext_vector_type(4))) unsigned int u32x4;
typedef __attribute__((ext_vector_type(8))) int          i32x8;
typedef __attribute__((ext_vector_type(4))) int          i32x4;

// 32-bit LDS byte offset of a __shared__ object (addrspacecast + ptrtoint)
__device__ __forceinline__ unsigned lds_off(const void* p) {
  return (unsigned)(unsigned long long)(const __attribute__((address_space(3))) char*)p;
}

// Issue a 2D TDM load: nrows rows of TK bf16, global row stride D_ elements,
// LDS destination padded to XPITCH (pad 4 DWORDs after every 16 DWORDs).
__device__ __forceinline__ void tdm_load_tile(unsigned ldsAddr, const void* gaddr,
                                              int nrows) {
  unsigned long long ga = (unsigned long long)gaddr;
  u32x4 g0 = {1u,                                  // count=1, no gather
              ldsAddr,                             // lds_addr
              (unsigned)(ga & 0xFFFFFFFFu),        // global_addr[31:0]
              (unsigned)((ga >> 32) & 0x1FFFFFFu) | (2u << 30)}; // hi | type=2
  i32x8 g1;
  g1[0] = (1 << 16)        // data_size = 1 -> 2 bytes
        | (1 << 20)        // pad_enable
        | (3 << 22)        // pad_interval code 3 -> every 16 DWORDs (one 64B row)
        | (3 << 25);       // pad_amount  code 3 -> 4 DWORDs (16B) => 80B pitch
  g1[1] = (TK & 0xFFFF) << 16;        // tensor_dim0[15:0]
  g1[2] = (nrows & 0xFFFF) << 16;     // tensor_dim0[31:16]=0 | tensor_dim1[15:0]
  g1[3] = (TK & 0xFFFF) << 16;        // tensor_dim1[31:16]=0 | tile_dim0
  g1[4] = nrows;                      // tile_dim1 | tile_dim2=0
  g1[5] = D_;                         // tensor_dim0_stride[31:0] (elements)
  g1[6] = 0;                          // stride0[47:32] | stride1[15:0]
  g1[7] = 0;                          // stride1[47:16]
  i32x4 z4 = {0, 0, 0, 0};
#if defined(__clang_major__) && (__clang_major__ >= 23)
  i32x8 z8 = {0, 0, 0, 0, 0, 0, 0, 0};
  __builtin_amdgcn_tensor_load_to_lds(g0, g1, z4, z4, z8, 0);
#else
  __builtin_amdgcn_tensor_load_to_lds(g0, g1, z4, z4, 0);
#endif
}
#endif  // USE_TDM

__device__ __forceinline__ float softplus_f(float x) {
  return log1pf(__expf(-fabsf(x))) + fmaxf(x, 0.0f);
}
__device__ __forceinline__ float logaddexp_f(float a, float b) {
  float m = fmaxf(a, b);
  if (m == -__builtin_inff()) return m;
  return m + log1pf(__expf(fminf(a, b) - m));
}

// ---------------------------------------------------------------------------
// Pre-pass 1: split x (f32) into bf16 hi + bf16 residual lo, natural layout.
// ---------------------------------------------------------------------------
__global__ __launch_bounds__(256) void mingru_split_x(
    const float* __restrict__ x, __bf16* __restrict__ xh, __bf16* __restrict__ xl)
{
  size_t i = ((size_t)blockIdx.x * 256 + threadIdx.x) * 4;
  float4 v = *(const float4*)(x + i);
  __bf16 h0 = (__bf16)v.x, h1 = (__bf16)v.y, h2 = (__bf16)v.z, h3 = (__bf16)v.w;
  v4bf hv = {h0, h1, h2, h3};
  v4bf lv = {(__bf16)(v.x - (float)h0), (__bf16)(v.y - (float)h1),
             (__bf16)(v.z - (float)h2), (__bf16)(v.w - (float)h3)};
  *(v4bf*)(xh + i) = hv;
  *(v4bf*)(xl + i) = lv;
}

// ---------------------------------------------------------------------------
// Pre-pass 2: transpose W (k,n)->(n,k) and split to bf16 hi/lo.
// ---------------------------------------------------------------------------
__global__ __launch_bounds__(256) void mingru_split_wT(
    const float* __restrict__ W, __bf16* __restrict__ wth, __bf16* __restrict__ wtl)
{
  __shared__ float tile[32][33];
  const int n0b = blockIdx.x * 32;
  const int k0  = blockIdx.y * 32;
  const int tx = threadIdx.x, ty = threadIdx.y;  // (32, 8)
  #pragma unroll
  for (int r = 0; r < 32; r += 8)
    tile[ty + r][tx] = W[(size_t)(k0 + ty + r) * N2_ + n0b + tx];
  __syncthreads();
  #pragma unroll
  for (int r = 0; r < 32; r += 8) {
    int n = n0b + ty + r;
    float v = tile[tx][ty + r];              // = W[k0+tx][n]
    __bf16 h = (__bf16)v;
    wth[(size_t)n * D_ + k0 + tx] = h;
    wtl[(size_t)n * D_ + k0 + tx] = (__bf16)(v - (float)h);
  }
}

// Fragment builders: contiguous bf16 runs in LDS per ISA 7.12.2 layouts.
__device__ __forceinline__ v16bf frag_a(const __bf16 (*Xs)[XPITCH], int row, int akb) {
  v8bf a0 = *(const v8bf*)&Xs[row][akb];
  v8bf a1 = *(const v8bf*)&Xs[row][akb + 16];
  return __builtin_shufflevector(a0, a1, 0,1,2,3,4,5,6,7,8,9,10,11,12,13,14,15);
}
__device__ __forceinline__ v16bf frag_b(const __bf16 (*Bs)[XPITCH], int col, int bkb) {
  v8bf b0 = *(const v8bf*)&Bs[col][bkb];
  v8bf b1 = *(const v8bf*)&Bs[col][bkb + 16];
  return __builtin_shufflevector(b0, b1, 0,1,2,3,4,5,6,7,8,9,10,11,12,13,14,15);
}

// ---------------------------------------------------------------------------
// Main kernel: bf16x3 WMMA GEMM (f32 accumulate) + fused log-space pointwise.
// Staging via Tensor Data Mover (TDM) when available: waves 0..5 each issue
// one 2D tile descriptor (Xh, Xl, Bh-hid, Bh-gate, Bl-hid, Bl-gate), wait on
// TENSORcnt, then the workgroup barrier publishes the LDS tiles.
// ---------------------------------------------------------------------------
__global__ __launch_bounds__(256) void mingru_gemm_pointwise(
    const __bf16* __restrict__ xh, const __bf16* __restrict__ xl,
    const __bf16* __restrict__ wth, const __bf16* __restrict__ wtl,
    float* __restrict__ logc, float* __restrict__ logv)
{
  __shared__ __align__(16) __bf16 Xh[TM][XPITCH];        // 128 x 40 bf16
  __shared__ __align__(16) __bf16 Xl[TM][XPITCH];
  __shared__ __align__(16) __bf16 Bh[2 * TNP][XPITCH];   // 128 cols (hid|gate) x 40
  __shared__ __align__(16) __bf16 Bl[2 * TNP][XPITCH];

  const int t    = threadIdx.x;
  const int lane = t & 31;
  const int wv   = t >> 5;       // 8 waves
  const int wm   = wv >> 2;      // 0..1 : 64-row half
  const int wn   = wv & 3;       // 0..3 : 16-col group
  const int mBase = blockIdx.x * TM;
  const int n0    = blockIdx.y * TNP;

  const int nLoc = lane & 15;
  const int akb  = (lane < 16) ? 0 : 8;    // A-fragment K base
  const int bkb  = (lane < 16) ? 0 : 8;    // B-fragment K base
  const int colH = wn * 16 + nLoc;
  const int colG = TNP + wn * 16 + nLoc;

#if USE_TDM
  // Per-wave DMA assignment (all values wave-uniform => SGPR descriptors)
  unsigned tdmLds = 0;
  const __bf16* tdmSrc = nullptr;
  int tdmRows = 0;
  if (wv == 0)      { tdmLds = lds_off(&Xh[0][0]);   tdmSrc = xh  + (size_t)mBase * D_;      tdmRows = TM;  }
  else if (wv == 1) { tdmLds = lds_off(&Xl[0][0]);   tdmSrc = xl  + (size_t)mBase * D_;      tdmRows = TM;  }
  else if (wv == 2) { tdmLds = lds_off(&Bh[0][0]);   tdmSrc = wth + (size_t)n0 * D_;         tdmRows = TNP; }
  else if (wv == 3) { tdmLds = lds_off(&Bh[TNP][0]); tdmSrc = wth + (size_t)(D_ + n0) * D_;  tdmRows = TNP; }
  else if (wv == 4) { tdmLds = lds_off(&Bl[0][0]);   tdmSrc = wtl + (size_t)n0 * D_;         tdmRows = TNP; }
  else if (wv == 5) { tdmLds = lds_off(&Bl[TNP][0]); tdmSrc = wtl + (size_t)(D_ + n0) * D_;  tdmRows = TNP; }
#endif

  v8f accH[4] = {{0}, {0}, {0}, {0}};
  v8f accG[4] = {{0}, {0}, {0}, {0}};

  for (int k0 = 0; k0 < D_; k0 += TK) {
#if USE_TDM
    if (wv < 6) {
      tdm_load_tile(tdmLds, tdmSrc + k0, tdmRows);
      __builtin_amdgcn_s_wait_tensorcnt(0);
    }
    __syncthreads();
#else
    // Fallback: synchronous b128 staging (128 rows x 32 bf16 per array)
    #pragma unroll
    for (int i = 0; i < 2; ++i) {
      int slot = t + i * 256;          // 0..511
      int row  = slot >> 2;            // 0..127
      int c    = (slot & 3) * 8;       // bf16 offset 0,8,16,24
      size_t xo = (size_t)(mBase + row) * D_ + k0 + c;
      *(uint4*)&Xh[row][c] = *(const uint4*)(xh + xo);
      *(uint4*)&Xl[row][c] = *(const uint4*)(xl + xo);
      int gcol = (row < TNP) ? (n0 + row) : (D_ + n0 + (row - TNP));
      size_t wo = (size_t)gcol * D_ + k0 + c;
      *(uint4*)&Bh[row][c] = *(const uint4*)(wth + wo);
      *(uint4*)&Bl[row][c] = *(const uint4*)(wtl + wo);
    }
    __syncthreads();
#endif

    const v16bf bHh = frag_b(Bh, colH, bkb);
    const v16bf bHl = frag_b(Bl, colH, bkb);
    const v16bf bGh = frag_b(Bh, colG, bkb);
    const v16bf bGl = frag_b(Bl, colG, bkb);

    #pragma unroll
    for (int i = 0; i < 4; ++i) {
      const int row = wm * 64 + i * 16 + nLoc;
      const v16bf ah = frag_a(Xh, row, akb);
      const v16bf al = frag_a(Xl, row, akb);
      accH[i] = __builtin_amdgcn_wmma_f32_16x16x32_bf16(false, ah, false, bHh,
                                                        (short)0, accH[i], false, false);
      accH[i] = __builtin_amdgcn_wmma_f32_16x16x32_bf16(false, ah, false, bHl,
                                                        (short)0, accH[i], false, false);
      accH[i] = __builtin_amdgcn_wmma_f32_16x16x32_bf16(false, al, false, bHh,
                                                        (short)0, accH[i], false, false);
      accG[i] = __builtin_amdgcn_wmma_f32_16x16x32_bf16(false, ah, false, bGh,
                                                        (short)0, accG[i], false, false);
      accG[i] = __builtin_amdgcn_wmma_f32_16x16x32_bf16(false, ah, false, bGl,
                                                        (short)0, accG[i], false, false);
      accG[i] = __builtin_amdgcn_wmma_f32_16x16x32_bf16(false, al, false, bGh,
                                                        (short)0, accG[i], false, false);
    }
    __syncthreads();
  }

  // ---- epilogue: log-space pointwise, write logc/logv in (b,s,d) layout
  const int d = n0 + wn * 16 + nLoc;
  #pragma unroll
  for (int i = 0; i < 4; ++i) {
    const int m0 = mBase + wm * 64 + i * 16 + ((lane < 16) ? 0 : 8);
    #pragma unroll
    for (int r = 0; r < 8; ++r) {
      float h = accH[i][r];
      float g = accG[i][r];
      float lc = -softplus_f(g);                  // log_coeffs
      float lz = -softplus_f(-g);                 // log_z
      float lg = (h >= 0.0f) ? __logf(h + 0.5f) : -softplus_f(-h);  // log_g(h)
      size_t off = (size_t)(m0 + r) * D_ + d;
      logc[off] = lc;
      logv[off] = lz + lg;
    }
  }
}

// ---------------------------------------------------------------------------
// Chunked log-space linear-recurrence scan: h_t = exp(c_t)*h_{t-1} + exp(v_t)
// ---------------------------------------------------------------------------
__global__ __launch_bounds__(256) void mingru_scan_pass1(
    const float* __restrict__ logc, const float* __restrict__ logv,
    float* __restrict__ cA, float* __restrict__ cB)
{
  int idx   = blockIdx.x * blockDim.x + threadIdx.x;  // (b*NC + chunk)*D + d
  int d     = idx & (D_ - 1);
  int chunk = (idx >> 10) & (NC - 1);
  int b     = idx >> 15;
  size_t base = ((size_t)b * S_ + (size_t)chunk * CHUNK) * D_ + d;
  float logA = 0.0f;
  float logB = -__builtin_inff();
  for (int s = 0; s < CHUNK; ++s) {
    float c = logc[base + (size_t)s * D_];
    float v = logv[base + (size_t)s * D_];
    logB = logaddexp_f(logB + c, v);
    logA += c;
  }
  cA[idx] = logA;
  cB[idx] = logB;
}

__global__ __launch_bounds__(256) void mingru_scan_pass2(
    const float* __restrict__ cA, const float* __restrict__ cB,
    float* __restrict__ carry)
{
  int idx = blockIdx.x * blockDim.x + threadIdx.x;    // b*D + d  (4096)
  int d   = idx & (D_ - 1);
  int b   = idx >> 10;
  float lH = -__builtin_inff();
  for (int c = 0; c < NC; ++c) {
    size_t o = ((size_t)(b * NC + c)) * D_ + d;
    carry[o] = lH;
    lH = logaddexp_f(cA[o] + lH, cB[o]);
  }
}

__global__ __launch_bounds__(256) void mingru_scan_pass3(
    const float* __restrict__ logc, const float* __restrict__ logv,
    const float* __restrict__ carry,
    float* __restrict__ out, float* __restrict__ lastH)
{
  int idx   = blockIdx.x * blockDim.x + threadIdx.x;
  int d     = idx & (D_ - 1);
  int chunk = (idx >> 10) & (NC - 1);
  int b     = idx >> 15;
  size_t base = ((size_t)b * S_ + (size_t)chunk * CHUNK) * D_ + d;
  float lH = carry[idx];
  for (int s = 0; s < CHUNK; ++s) {
    float c = logc[base + (size_t)s * D_];
    float v = logv[base + (size_t)s * D_];
    lH = logaddexp_f(lH + c, v);
    out[base + (size_t)s * D_] = __expf(lH);
  }
  if (chunk == NC - 1) {
    lastH[(size_t)b * D_ + d] = __expf(lH);   // next_prev_hidden
  }
}

// ---------------------------------------------------------------------------
extern "C" void kernel_launch(void* const* d_in, const int* in_sizes, int n_in,
                              void* d_out, int out_size, void* d_ws, size_t ws_size,
                              hipStream_t stream) {
  const float* x = (const float*)d_in[0];   // (B, S, D) f32
  const float* W = (const float*)d_in[1];   // (D, 2D) f32
  float* out = (float*)d_out;               // (B,S,D) then (B,1,D)

  // Workspace layout (f32 blocks first, then bf16 blocks; all 16B aligned)
  float* logc  = (float*)d_ws;
  float* logv  = logc + (size_t)M_ * D_;
  float* cA    = logv + (size_t)M_ * D_;
  float* cB    = cA + (size_t)B_ * NC * D_;
  float* carry = cB + (size_t)B_ * NC * D_;
  __bf16* xh  = (__bf16*)(carry + (size_t)B_ * NC * D_);
  __bf16* xl  = xh + (size_t)M_ * D_;
  __bf16* wth = xl + (size_t)M_ * D_;
  __bf16* wtl = wth + (size_t)N2_ * D_;

  // Pre-split (bandwidth bound, ~12us total at 23.3 TB/s)
  mingru_split_x<<<(M_ * (size_t)D_) / (256 * 4), 256, 0, stream>>>(x, xh, xl);
  mingru_split_wT<<<dim3(N2_ / 32, D_ / 32), dim3(32, 8), 0, stream>>>(W, wth, wtl);

  // WMMA GEMM + fused pointwise
  dim3 gGemm(M_ / TM, D_ / TNP);            // 256 x 16 workgroups
  mingru_gemm_pointwise<<<gGemm, 256, 0, stream>>>(xh, xl, wth, wtl, logc, logv);

  // Chunked scan
  int scanThreads = B_ * NC * D_;           // 131072
  mingru_scan_pass1<<<scanThreads / 256, 256, 0, stream>>>(logc, logv, cA, cB);
  mingru_scan_pass2<<<(B_ * D_) / 256, 256, 0, stream>>>(cA, cB, carry);
  mingru_scan_pass3<<<scanThreads / 256, 256, 0, stream>>>(
      logc, logv, carry, out, out + (size_t)M_ * D_);
}
